// GeodesicLMHeadV2_33088428049141
// MI455X (gfx1250) — compile-verified, matrix operand order checked by reading, not run on hardware
//
#include <hip/hip_runtime.h>

typedef __attribute__((ext_vector_type(16))) __bf16 v16bf;
typedef __attribute__((ext_vector_type(8)))  float  v8f;

#define M_ROWS 4096     // B*L
#define K_DIM  1024     // n_embd
#define V_DIM  32000    // vocab
#define AMB    1025     // N+1
#define KSTEP  32
#define NSTEPS (K_DIM / KSTEP)   // 32

__device__ __forceinline__ float sanitize_f(float x) {
    float s = (x != x) ? 0.0f : x;                 // NaN -> 0
    if (isinf(s)) s = (s > 0.0f) ? 1e4f : -1e4f;   // +/-inf -> +/-1e4
    return s;
}

__device__ __forceinline__ float safe_acosh_f(float x) {
    float xm1    = fmaxf(x - 1.0f, 0.0f);
    float taylor = sqrtf(2.0f * fmaxf(xm1, 1e-12f)) * (1.0f - xm1 * (1.0f / 12.0f));
    float xc     = fmaxf(x, 1.0f + 1e-7f);
    float exact  = __logf(xc + sqrtf(xc * xc - 1.0f));   // hardware v_log_f32
    return (xm1 < 1e-3f) ? taylor : exact;
}

// ---------------- prep: hidden states -> bf16 hi/lo split + time component ----
__global__ void pack_hidden(const float* __restrict__ h,
                            __bf16* __restrict__ Ahi, __bf16* __restrict__ Alo,
                            float* __restrict__ htime) {
    const int m = blockIdx.x;
    const float* row = h + (size_t)m * AMB;
    if (threadIdx.x == 0) htime[m] = sanitize_f(__builtin_nontemporal_load(&row[0]));
    for (int k = threadIdx.x; k < K_DIM; k += blockDim.x) {
        float v  = sanitize_f(__builtin_nontemporal_load(&row[1 + k]));
        __bf16 hi = (__bf16)v;
        float  lo = v - (float)hi;
        Ahi[(size_t)m * K_DIM + k] = hi;
        Alo[(size_t)m * K_DIM + k] = (__bf16)lo;
    }
}

// ---------------- prep: tangent weights -> exp-map + bf16 hi/lo split ---------
__global__ void pack_weight(const float* __restrict__ w,
                            __bf16* __restrict__ Whi, __bf16* __restrict__ Wlo,
                            float* __restrict__ wtime) {
    __shared__ float red[256];
    const int v = blockIdx.x;
    const float* row = w + (size_t)v * K_DIM;
    float s = 0.0f;
    for (int k = threadIdx.x; k < K_DIM; k += blockDim.x) {
        float x = row[k];
        s += x * x;
    }
    red[threadIdx.x] = s;
    __syncthreads();
    for (int off = 128; off > 0; off >>= 1) {
        if ((int)threadIdx.x < off) red[threadIdx.x] += red[threadIdx.x + off];
        __syncthreads();
    }
    float norm = sqrtf(red[0]);
    float r    = (norm > 1e-7f) ? (sinhf(norm) / fmaxf(norm, 1e-7f)) : 1.0f;
    if (threadIdx.x == 0) wtime[v] = coshf(norm);
    for (int k = threadIdx.x; k < K_DIM; k += blockDim.x) {
        float x   = r * row[k];
        __bf16 hi = (__bf16)x;
        Whi[(size_t)v * K_DIM + k] = hi;
        Wlo[(size_t)v * K_DIM + k] = (__bf16)(x - (float)hi);
    }
}

// ---------------- main GEMM + geodesic epilogue -------------------------------
// grid = (V/128, M/128); block = 256 threads = 8 waves arranged 2(M) x 4(N);
// wave tile = 64(M) x 32(N) = 4x2 fragments, K-step 32.
// Tiles staged in LDS via async DMA (ASYNCcnt), double-buffered:
//   stage layout: [2 stages][4 planes: Ahi,Alo,Bhi,Blo][128 rows][64 bytes]
//   16B granules XOR-swizzled by (row>>2)&3 to spread LDS banks.
// x = A_hi*B_hi + A_hi*B_lo + A_lo*B_hi  (3 bf16 WMMAs, fp32 accum)
#define PLANE_BYTES (128 * 64)
#define STAGE_BYTES (4 * PLANE_BYTES)

__global__ void __launch_bounds__(256, 1)
geo_gemm(const __bf16* __restrict__ Ahi, const __bf16* __restrict__ Alo,
         const __bf16* __restrict__ Whi, const __bf16* __restrict__ Wlo,
         const float* __restrict__ htime, const float* __restrict__ wtime,
         const float* __restrict__ lscale, float* __restrict__ out) {
    __shared__ __align__(16) char smem[2 * STAGE_BYTES];   // 64 KB

    const int lane   = threadIdx.x & 31;
    const int waveId = threadIdx.x >> 5;
    const int lane16 = lane & 15;
    const int laneHi = lane >> 4;
    const int bm0 = blockIdx.y * 128;
    const int bn0 = blockIdx.x * 128;
    const int wmL = (waveId & 1) * 64;   // wave row offset in block (M)
    const int wnL = (waveId >> 1) * 32;  // wave col offset in block (N)
    const int wm0 = bm0 + wmL;
    const int wn0 = bn0 + wnL;

    // --- async fill assignment: each wave owns one plane, 64 rows, 8 chunks ---
    const char* planeBase[4] = {(const char*)Ahi, (const char*)Alo,
                                (const char*)Whi, (const char*)Wlo};
    const int   fillPlane = waveId >> 1;               // 0..3
    const int   fillRow0  = (waveId & 1) * 64;         // 0 or 64
    const char* fillBase  = planeBase[fillPlane];
    const int   fillTile0 = (fillPlane < 2) ? bm0 : bn0;
    const int   frow = lane >> 2;                      // 0..7 within a chunk
    const int   fsub = lane & 3;                       // 16B granule in row

    const uint32_t smem0 = (uint32_t)(uintptr_t)&smem[0];
    const uint32_t fillPlaneOff = (uint32_t)fillPlane * PLANE_BYTES;

    auto fill_stage = [&](int stage, int k) {
        const uint32_t sbase = smem0 + (uint32_t)stage * STAGE_BYTES + fillPlaneOff;
#pragma unroll
        for (int c = 0; c < 8; ++c) {
            const int row = fillRow0 + c * 8 + frow;
            const uint32_t goff =
                (uint32_t)((fillTile0 + row) * (K_DIM * 2) + k * 2 + fsub * 16);
            const uint32_t loff =
                sbase + (uint32_t)(row * 64 + ((fsub ^ ((row >> 2) & 3)) * 16));
            asm volatile("global_load_async_to_lds_b128 %0, %1, %2"
                         :: "v"(loff), "v"(goff), "s"(fillBase)
                         : "memory");
        }
    };

    auto lds_frag = [&](int stage, int plane, int row) -> v16bf {
        const char* p = &smem[0] + stage * STAGE_BYTES + plane * PLANE_BYTES
                        + row * 64;
        const int swz = (row >> 2) & 3;
        union { v16bf v; uint4 q[2]; } f;
        f.q[0] = *(const uint4*)(p + ((laneHi ^ swz) * 16));
        f.q[1] = *(const uint4*)(p + (((laneHi + 2) ^ swz) * 16));
        return f.v;
    };

    v8f acc[4][2] = {};

    fill_stage(0, 0);   // prologue

    for (int t = 0; t < NSTEPS; ++t) {
        if (t + 1 < NSTEPS) {
            fill_stage((t + 1) & 1, (t + 1) * KSTEP);
            // previous stage's 8 async loads done (in-order), next 8 in flight
            asm volatile("s_wait_asynccnt 0x8" ::: "memory");
        } else {
            asm volatile("s_wait_asynccnt 0x0" ::: "memory");
        }
        __syncthreads();

        const int st = t & 1;
        v16bf ah[4], al[4], bh[2], bl[2];
#pragma unroll
        for (int i = 0; i < 4; ++i) {
            ah[i] = lds_frag(st, 0, wmL + 16 * i + lane16);
            al[i] = lds_frag(st, 1, wmL + 16 * i + lane16);
        }
#pragma unroll
        for (int j = 0; j < 2; ++j) {
            bh[j] = lds_frag(st, 2, wnL + 16 * j + lane16);
            bl[j] = lds_frag(st, 3, wnL + 16 * j + lane16);
        }
#pragma unroll
        for (int i = 0; i < 4; ++i) {
#pragma unroll
            for (int j = 0; j < 2; ++j) {
                acc[i][j] = __builtin_amdgcn_wmma_f32_16x16x32_bf16(
                    false, ah[i], false, bh[j], (short)0, acc[i][j], false, false);
                acc[i][j] = __builtin_amdgcn_wmma_f32_16x16x32_bf16(
                    false, ah[i], false, bl[j], (short)0, acc[i][j], false, false);
                acc[i][j] = __builtin_amdgcn_wmma_f32_16x16x32_bf16(
                    false, al[i], false, bh[j], (short)0, acc[i][j], false, false);
            }
        }
        __syncthreads();   // all waves done reading stage st before it refills
    }

    const float tau = fminf(fmaxf(lscale[0], 0.01f), 2.5f);

#pragma unroll
    for (int i = 0; i < 4; ++i) {
        float ht[8];
#pragma unroll
        for (int r = 0; r < 8; ++r)
            ht[r] = htime[wm0 + 16 * i + r + 8 * laneHi];
#pragma unroll
        for (int j = 0; j < 2; ++j) {
            const int   gn = wn0 + 16 * j + lane16;
            const float w0 = wtime[gn];
#pragma unroll
            for (int r = 0; r < 8; ++r) {
                const int gm = wm0 + 16 * i + r + 8 * laneHi;
                float x = ht[r] * w0 - acc[i][j][r];   // -<h,w>_L
                float d = safe_acosh_f(x);
                float lg = sanitize_f(-tau * d * d);
                // streaming store: keep 512MB output out of L2 so the bf16
                // weight slab (131MB) stays resident
                __builtin_nontemporal_store(lg, &out[(size_t)gm * V_DIM + gn]);
            }
        }
    }
}

// ---------------- host launcher ----------------------------------------------
extern "C" void kernel_launch(void* const* d_in, const int* in_sizes, int n_in,
                              void* d_out, int out_size, void* d_ws, size_t ws_size,
                              hipStream_t stream) {
    const float* h  = (const float*)d_in[0];   // [2, 2048, 1025]
    const float* w  = (const float*)d_in[1];   // [32000, 1024]
    const float* ls = (const float*)d_in[2];   // scalar
    float* out = (float*)d_out;                // [2, 2048, 32000] fp32

    char* ws = (char*)d_ws;
    size_t offAhi = 0;
    size_t offAlo = offAhi + (size_t)M_ROWS * K_DIM * 2;
    size_t offWhi = offAlo + (size_t)M_ROWS * K_DIM * 2;
    size_t offWlo = offWhi + (size_t)V_DIM * K_DIM * 2;
    size_t offHt  = offWlo + (size_t)V_DIM * K_DIM * 2;
    size_t offWt  = offHt  + (size_t)M_ROWS * 4;

    __bf16* Ahi = (__bf16*)(ws + offAhi);
    __bf16* Alo = (__bf16*)(ws + offAlo);
    __bf16* Whi = (__bf16*)(ws + offWhi);
    __bf16* Wlo = (__bf16*)(ws + offWlo);
    float*  Ht  = (float*)(ws + offHt);
    float*  Wt  = (float*)(ws + offWt);

    pack_hidden<<<M_ROWS, 256, 0, stream>>>(h, Ahi, Alo, Ht);
    pack_weight<<<V_DIM, 256, 0, stream>>>(w, Whi, Wlo, Wt);

    dim3 grid(V_DIM / 128, M_ROWS / 128);   // 250 x 32
    geo_gemm<<<grid, 256, 0, stream>>>(Ahi, Alo, Whi, Wlo, Ht, Wt, ls, out);
}
